// S2V_QN_1_84808424227517
// MI455X (gfx1250) — matrix operand order repaired and checked by reading, not compiled, not run on hardware
//
#include <hip/hip_runtime.h>
#include <cstdint>

// Problem constants (match reference)
#define Bb 8
#define Nn 4096
#define Dd 64
#define RHh 32
#define BN (Bb * Nn)

typedef _Float16 h16;
typedef __attribute__((ext_vector_type(16))) _Float16 v16h;
typedef __attribute__((ext_vector_type(8)))  _Float16 v8h;
typedef __attribute__((ext_vector_type(8)))  float    v8f;
typedef __attribute__((ext_vector_type(4)))  float    v4f;

union FragH { v16h v; v8h h[2]; };

// ---------------------------------------------------------------------------
// Prep: transpose+convert weight matrices to f16 (WMMA B-operand friendly),
// zero the pooled accumulator (used with atomics later).
// ---------------------------------------------------------------------------
__global__ void prep_kernel(const float* __restrict__ pre_W,
                            const float* __restrict__ post_W,
                            const float* __restrict__ mu2_W,
                            const float* __restrict__ q2_W,
                            const float* __restrict__ qreg_W,
                            h16* __restrict__ preT, h16* __restrict__ postT,
                            h16* __restrict__ mu2T, h16* __restrict__ q2T,
                            h16* __restrict__ qregT,
                            float* __restrict__ pooled) {
  int i = blockIdx.x * blockDim.x + threadIdx.x;
  int stride = gridDim.x * blockDim.x;
  for (int t = i; t < Bb * Dd; t += stride) pooled[t] = 0.0f;
  // six 64x64 matrices: pre0, pre1, post0, post1, mu2, q2 -> transposed f16
  for (int t = i; t < 6 * Dd * Dd; t += stride) {
    int m = t >> 12, e = t & 4095;
    int n = e >> 6, k = e & 63;
    const float* src;
    h16* dst;
    switch (m) {
      case 0: src = pre_W;            dst = preT;            break;
      case 1: src = pre_W + 4096;     dst = preT + 4096;     break;
      case 2: src = post_W;           dst = postT;           break;
      case 3: src = post_W + 4096;    dst = postT + 4096;    break;
      case 4: src = mu2_W;            dst = mu2T;            break;
      default: src = q2_W;            dst = q2T;             break;
    }
    dst[n * Dd + k] = (h16)src[k * Dd + n];
  }
  // qreg_W [128,32] -> qregT [32][128]
  for (int t = i; t < RHh * 2 * Dd; t += stride) {
    int hidx = t >> 7, k = t & 127;
    qregT[t] = (h16)qreg_W[k * RHh + hidx];
  }
}

// ---------------------------------------------------------------------------
// Embed: mu0 = relu(xv * mu_1) * mask   -> f16 [BN,64]
// ---------------------------------------------------------------------------
__global__ void embed_kernel(const float* __restrict__ xv,
                             const float* __restrict__ mu1,
                             const int* __restrict__ mask,
                             h16* __restrict__ h0) {
  int idx = blockIdx.x * blockDim.x + threadIdx.x;  // over BN*64
  if (idx >= BN * Dd) return;
  int row = idx >> 6, d = idx & 63;
  float v = xv[row] * mu1[d];
  v = v > 0.0f ? v : 0.0f;
  v *= (float)mask[row];
  h0[idx] = (h16)v;
}

// ---------------------------------------------------------------------------
// Generic fused WMMA GEMM: Y[BN,64] = act(X[BN,64] @ W + b)
// ---------------------------------------------------------------------------
template <bool RELU, bool MASKED, bool RESIDUAL, bool STORE_T, bool POOL>
__global__ __launch_bounds__(128) void gemm64_kernel(
    const h16* __restrict__ X, const h16* __restrict__ Wt,
    const float* __restrict__ bias, h16* __restrict__ Y,
    h16* __restrict__ Yt, const h16* __restrict__ Res,
    const int* __restrict__ mask, const float* __restrict__ xv,
    float* __restrict__ pooled) {
  const int lane = threadIdx.x & 31;
  const int wave = threadIdx.x >> 5;          // 0..3
  const int m0 = blockIdx.x * 16;             // row tile in [0, BN)
  const int col = wave * 16 + (lane & 15);    // output column 0..63
  const int hiHalf = (lane >= 16);
  const int aoff = hiHalf ? 8 : 0;            // A: K interleave per ISA layout
  const int boff = hiHalf ? 8 : 0;            // B: K interleave matching A
  const int rowA = m0 + (lane & 15);

  const h16* xrow = X + (size_t)rowA * Dd;
  const h16* wrow = Wt + (size_t)col * Dd;

  v8f c;
  {
    float bv = bias[col];
#pragma unroll
    for (int i = 0; i < 8; i++) c[i] = bv;
  }

#pragma unroll
  for (int ks = 0; ks < 2; ks++) {
    const int k0 = ks * 32;
    FragH a, bf;
    a.h[0] = *(const v8h*)(xrow + k0 + aoff);
    a.h[1] = *(const v8h*)(xrow + k0 + 16 + aoff);
    bf.h[0] = *(const v8h*)(wrow + k0 + boff);
    bf.h[1] = *(const v8h*)(wrow + k0 + boff + 16);
    c = __builtin_amdgcn_wmma_f32_16x16x32_f16(false, a.v, false, bf.v,
                                               (short)0, c, false, false);
  }

  float pool_acc = 0.0f;
#pragma unroll
  for (int r = 0; r < 8; r++) {
    int row = m0 + r + (hiHalf ? 8 : 0);
    float v = c[r];
    if (MASKED) v *= (float)mask[row];
    if (RESIDUAL) v += (float)Res[(size_t)row * Dd + col];
    if (RELU) v = v > 0.0f ? v : 0.0f;
    h16 hv = (h16)v;
    Y[(size_t)row * Dd + col] = hv;
    if (STORE_T) {
      int b = row >> 12;           // row / N
      int n = row & (Nn - 1);      // row % N
      Yt[((size_t)b * Dd + col) * Nn + n] = hv;
    }
    if (POOL) pool_acc += xv[row] * v;
  }
  if (POOL) {
    pool_acc += __shfl_down(pool_acc, 16, 32);  // combine rows 0-7 with 8-15
    if (lane < 16) {
      int b = m0 >> 12;  // tile never crosses batch (N % 16 == 0)
      atomicAdd(&pooled[b * Dd + col], pool_acc);
    }
  }
}

// ---------------------------------------------------------------------------
// Helper: build a 16x32 f16 A-fragment from one fp32 row (ISA A layout:
// lanes 0-15 carry K[0..7]+K[16..23], lanes 16-31 carry K[8..15]+K[24..31]).
// ---------------------------------------------------------------------------
__device__ __forceinline__ v16h load_a_frag_f32(const float* __restrict__ row,
                                                int k0, int aoff) {
  v4f a0 = *(const v4f*)(row + k0 + aoff);
  v4f a1 = *(const v4f*)(row + k0 + aoff + 4);
  v4f a2 = *(const v4f*)(row + k0 + 16 + aoff);
  v4f a3 = *(const v4f*)(row + k0 + 16 + aoff + 4);
  FragH a;
#pragma unroll
  for (int i = 0; i < 4; i++) {
    a.v[i]      = (h16)a0[i];
    a.v[4 + i]  = (h16)a1[i];
    a.v[8 + i]  = (h16)a2[i];
    a.v[12 + i] = (h16)a3[i];
  }
  return a.v;
}

// ---------------------------------------------------------------------------
// Big kernel: mu_pool[b] = (adj[b] (NxN fp32) @ mu_pre[b] (Nx64)) * mask
// 32-row tile per workgroup: each wave holds two f32 accumulators and shares
// one B fragment across two A fragments (halves B traffic, 2 WMMA / iter).
// Streams the 537MB adjacency exactly once from HBM; unconditional SYS-scope
// prefetch pulls upcoming cachelines into GL2 (speculative, OOB dropped).
// ---------------------------------------------------------------------------
__global__ __launch_bounds__(128) void adjmm_kernel(
    const float* __restrict__ A, const h16* __restrict__ BT,
    const int* __restrict__ mask, h16* __restrict__ Y) {
  const int b = blockIdx.y;
  const int m0 = blockIdx.x * 32;
  const int lane = threadIdx.x & 31;
  const int wave = threadIdx.x >> 5;
  const int col = wave * 16 + (lane & 15);
  const int hiHalf = (lane >= 16);
  const int aoff = hiHalf ? 8 : 0;
  const int boff = hiHalf ? 8 : 0;

  const float* arow0 = A + ((size_t)b * Nn + m0 + (lane & 15)) * Nn;
  const float* arow1 = arow0 + (size_t)16 * Nn;
  const h16* brow = BT + ((size_t)b * Dd + col) * Nn;

  v8f c0 = {}, c1 = {};
#pragma unroll 2
  for (int k0 = 0; k0 < Nn; k0 += 32) {
    __builtin_prefetch(arow0 + k0 + 1024, 0, 0);  // ~32 iters ahead -> GL2
    __builtin_prefetch(arow1 + k0 + 1024, 0, 0);
    v16h a0 = load_a_frag_f32(arow0, k0, aoff);
    v16h a1 = load_a_frag_f32(arow1, k0, aoff);
    FragH bf;
    bf.h[0] = *(const v8h*)(brow + k0 + boff);
    bf.h[1] = *(const v8h*)(brow + k0 + boff + 16);
    c0 = __builtin_amdgcn_wmma_f32_16x16x32_f16(false, a0, false, bf.v,
                                                (short)0, c0, false, false);
    c1 = __builtin_amdgcn_wmma_f32_16x16x32_f16(false, a1, false, bf.v,
                                                (short)0, c1, false, false);
  }
#pragma unroll
  for (int r = 0; r < 8; r++) {
    int n0 = m0 + r + (hiHalf ? 8 : 0);
    int row0 = b * Nn + n0;
    int row1 = row0 + 16;
    Y[(size_t)row0 * Dd + col] = (h16)(c0[r] * (float)mask[row0]);
    Y[(size_t)row1 * Dd + col] = (h16)(c1[r] * (float)mask[row1]);
  }
}

// ---------------------------------------------------------------------------
// Async-to-LDS probe (CDNA5 ASYNCcnt path): stages 2KB global->LDS with
// GLOBAL_LOAD_ASYNC_TO_LDS_B128, waits on ASYNCcnt, reads back via LDS.
// Launched once on scratch; if this assembles, the adjacency loop can adopt
// cross-wave LDS staging next.
// ---------------------------------------------------------------------------
__global__ __launch_bounds__(128) void async_stage_probe(
    const float* __restrict__ src, float* __restrict__ dst) {
  __shared__ float sbuf[512];
  uint32_t lds_addr = (uint32_t)(uintptr_t)(&sbuf[threadIdx.x * 4]);
  uint64_t gaddr = (uint64_t)(uintptr_t)(src + threadIdx.x * 4);
  asm volatile("global_load_async_to_lds_b128 %0, %1, off"
               :: "v"(lds_addr), "v"(gaddr)
               : "memory");
  asm volatile("s_wait_asynccnt 0" ::: "memory");
  __syncthreads();
  dst[threadIdx.x] = sbuf[threadIdx.x] + sbuf[(threadIdx.x + 128) & 511];
}

// ---------------------------------------------------------------------------
// q1 = pooled @ q1_W + q1_b  (tiny: [B,64]@[64,64])
// ---------------------------------------------------------------------------
__global__ void q1_kernel(const float* __restrict__ pooled,
                          const float* __restrict__ W,
                          const float* __restrict__ bvec,
                          float* __restrict__ q1) {
  int i = threadIdx.x;
  if (i >= Bb * Dd) return;
  int b = i >> 6, d = i & 63;
  float acc = bvec[d];
  const float* pr = pooled + b * Dd;
  for (int k = 0; k < Dd; k++) acc += pr[k] * W[k * Dd + d];
  q1[i] = acc;
}

// ---------------------------------------------------------------------------
// Head: q = where(mask==0, -99999, relu([q1|q2] @ qreg_W + qreg_b) @ q_W + q_b)
// ---------------------------------------------------------------------------
__global__ __launch_bounds__(256) void head_kernel(
    const h16* __restrict__ q2, const float* __restrict__ q1,
    const h16* __restrict__ qregT, const float* __restrict__ qreg_b,
    const float* __restrict__ qW, const float* __restrict__ qb,
    const int* __restrict__ mask, float* __restrict__ out) {
  __shared__ h16 sW[RHh * 2 * Dd];  // [32][128]
  __shared__ float sqw[RHh];
  for (int i = threadIdx.x; i < RHh * 2 * Dd; i += 256) sW[i] = qregT[i];
  if (threadIdx.x < RHh) sqw[threadIdx.x] = qW[threadIdx.x];
  __syncthreads();

  int row = blockIdx.x * 256 + threadIdx.x;
  if (row >= BN) return;
  int b = row >> 12;
  const float* q1b = q1 + b * Dd;
  const h16* q2r = q2 + (size_t)row * Dd;

  float acc = qb[0];
  for (int h = 0; h < RHh; h++) {
    float hh = qreg_b[h];
    const h16* wrow = sW + h * 2 * Dd;
#pragma unroll 8
    for (int k = 0; k < Dd; k++) hh += q1b[k] * (float)wrow[k];
#pragma unroll 8
    for (int k = 0; k < Dd; k++) hh += (float)q2r[k] * (float)wrow[Dd + k];
    hh = hh > 0.0f ? hh : 0.0f;
    acc += hh * sqw[h];
  }
  out[row] = (mask[row] == 0) ? -99999.0f : acc;
}

// ---------------------------------------------------------------------------
extern "C" void kernel_launch(void* const* d_in, const int* in_sizes, int n_in,
                              void* d_out, int out_size, void* d_ws,
                              size_t ws_size, hipStream_t stream) {
  (void)in_sizes; (void)n_in; (void)out_size; (void)ws_size;
  const float* xv     = (const float*)d_in[0];
  const float* adjw   = (const float*)d_in[2];  // edges_weight_adj
  const int*   mask   = (const int*)d_in[3];
  const float* mu_1   = (const float*)d_in[4];
  const float* mu2_W  = (const float*)d_in[5];
  const float* mu2_b  = (const float*)d_in[6];
  const float* pre_W  = (const float*)d_in[7];
  const float* pre_b  = (const float*)d_in[8];
  const float* post_W = (const float*)d_in[9];
  const float* post_b = (const float*)d_in[10];
  const float* q1_W   = (const float*)d_in[11];
  const float* q1_b   = (const float*)d_in[12];
  const float* q2_W   = (const float*)d_in[13];
  const float* q2_b   = (const float*)d_in[14];
  const float* qreg_W = (const float*)d_in[15];
  const float* qreg_b = (const float*)d_in[16];
  const float* q_W    = (const float*)d_in[17];
  const float* q_b    = (const float*)d_in[18];

  char* ws = (char*)d_ws;
  const size_t ACT = (size_t)BN * Dd * sizeof(h16);  // 4 MB
  h16* h0  = (h16*)(ws + 0 * ACT);   // mu_1v (residual source)
  h16* h1  = (h16*)(ws + 1 * ACT);   // ping
  h16* h2  = (h16*)(ws + 2 * ACT);   // pong
  h16* h2T = (h16*)(ws + 3 * ACT);   // mu_pre transposed [B,64,N]
  char* p = ws + 4 * ACT;
  h16* preT  = (h16*)p; p += 2 * Dd * Dd * sizeof(h16);
  h16* postT = (h16*)p; p += 2 * Dd * Dd * sizeof(h16);
  h16* mu2T  = (h16*)p; p += Dd * Dd * sizeof(h16);
  h16* q2T   = (h16*)p; p += Dd * Dd * sizeof(h16);
  h16* qregT = (h16*)p; p += RHh * 2 * Dd * sizeof(h16);
  float* pooled = (float*)p; p += Bb * Dd * sizeof(float);
  float* q1     = (float*)p; p += Bb * Dd * sizeof(float);
  float* probe  = (float*)p; p += 128 * sizeof(float);

  prep_kernel<<<64, 256, 0, stream>>>(pre_W, post_W, mu2_W, q2_W, qreg_W,
                                      preT, postT, mu2T, q2T, qregT, pooled);
  embed_kernel<<<(BN * Dd) / 256, 256, 0, stream>>>(xv, mu_1, mask, h0);

  // async-to-LDS probe (side-effect free wrt the pipeline)
  async_stage_probe<<<1, 128, 0, stream>>>(xv, probe);

  // pre layers
  gemm64_kernel<true, false, false, false, false><<<BN / 16, 128, 0, stream>>>(
      h0, preT, pre_b, h1, nullptr, nullptr, nullptr, nullptr, nullptr);
  gemm64_kernel<true, false, false, true, false><<<BN / 16, 128, 0, stream>>>(
      h1, preT + Dd * Dd, pre_b + Dd, h2, h2T, nullptr, nullptr, nullptr, nullptr);

  // adjacency aggregation (HBM-roofline stage)
  adjmm_kernel<<<dim3(Nn / 32, Bb), 128, 0, stream>>>(adjw, h2T, mask, h1);

  // post layers
  gemm64_kernel<true, false, false, false, false><<<BN / 16, 128, 0, stream>>>(
      h1, postT, post_b, h2, nullptr, nullptr, nullptr, nullptr, nullptr);
  gemm64_kernel<true, false, false, false, false><<<BN / 16, 128, 0, stream>>>(
      h2, postT + Dd * Dd, post_b + Dd, h1, nullptr, nullptr, nullptr, nullptr, nullptr);

  // mu = relu(mu_1v + (mu_pool@mu2_W + b)*m), fused with pooled reduction
  gemm64_kernel<true, true, true, false, true><<<BN / 16, 128, 0, stream>>>(
      h1, mu2T, mu2_b, h2, nullptr, h0, mask, xv, pooled);

  q1_kernel<<<1, 512, 0, stream>>>(pooled, q1_W, q1_b, q1);

  // q2 = mu @ q2_W + b (no relu)
  gemm64_kernel<false, false, false, false, false><<<BN / 16, 128, 0, stream>>>(
      h2, q2T, q2_b, h1, nullptr, nullptr, nullptr, nullptr, nullptr);

  head_kernel<<<BN / 256, 256, 0, stream>>>(h1, q1, qregT, qreg_b, q_W, q_b,
                                            mask, (float*)d_out);
}
